// MLAttention_54485955117115
// MI455X (gfx1250) — matrix operand order repaired and checked by reading, not compiled, not run on hardware
//
#include <hip/hip_runtime.h>
#include <hip/hip_bf16.h>

// MLA forward for gfx1250 (MI455X).
// - All matmuls through v_wmma_f32_16x16x32_bf16 (f32 WMMA is 16x16x4 -> 8x less K/instr,
//   so bf16-in/f32-acc is the throughput-correct precision on this chip).
// - One-time bf16 conversion of x + transposed bf16 weights => every GEMM/attention LDS
//   stage is a pure contiguous copy => GLOBAL_LOAD_ASYNC_TO_LDS_B128 (ASYNCcnt-tracked)
//   with LDS double buffering.
// - Flash attention computes S^T = K*Q^T so softmax is per-lane + one shfl_xor(16),
//   and P^T is repacked in-register into the WMMA B layout for O^T += V^T * P^T.

#define NH_   16
#define HD_   128
#define DROPE 64
#define DNOPE 64
#define DC    512
#define HDIM  2048
#define BB    2
#define LL    2048
#define NT    (BB * LL)   // 4096 tokens
#define SCALE 0.08838834764831845f   // 1/sqrt(128)
#define NEG_BIG (-3.0e38f)

typedef __attribute__((ext_vector_type(16))) __bf16 bf16x16;
typedef __attribute__((ext_vector_type(8)))  float  f32x8;

union Frag {
    bf16x16        v;
    unsigned int   u32[8];
    unsigned short u16[16];
};

__device__ __forceinline__ unsigned short f2bf(float f) {
    union { float f; unsigned u; } x; x.f = f;
    unsigned r = (x.u + 0x7FFFu + ((x.u >> 16) & 1u)) >> 16;  // RNE
    return (unsigned short)r;
}
__device__ __forceinline__ float bf2f(unsigned short b) {
    union { unsigned u; float f; } x; x.u = ((unsigned)b) << 16;
    return x.f;
}
__device__ __forceinline__ f32x8 zero8() {
    f32x8 v;
#pragma unroll
    for (int i = 0; i < 8; ++i) v[i] = 0.f;
    return v;
}
__device__ __forceinline__ f32x8 wmma_bf16(bf16x16 a, bf16x16 b, f32x8 c) {
    return __builtin_amdgcn_wmma_f32_16x16x32_bf16(false, a, false, b, (short)0, c, false, false);
}
// A-operand (16x32, 16-bit): pair j -> K = 2j + 16*(lane>=16)
__device__ __forceinline__ int fragA_k(int j, int hi) { return 2 * j + (hi ? 16 : 0); }
// B-operand (32x16, 16-bit): pair j -> K = 2j + 16*(lane>=16)
__device__ __forceinline__ int fragB_k(int j, int hi) { return 2 * j + (hi ? 16 : 0); }

// ---------------- async global->LDS copy path (guarded; falls back to uint4 copies) ----
#if defined(__has_builtin)
#if __has_builtin(__builtin_amdgcn_global_load_async_to_lds_b128) && __has_builtin(__builtin_amdgcn_s_wait_asynccnt)
#define HAVE_ASYNC_LDS 1
#endif
#endif
#ifndef HAVE_ASYNC_LDS
#define HAVE_ASYNC_LDS 0
#endif

#if HAVE_ASYNC_LDS
// param 1 per hipcc diagnostic: pointer to __attribute__((vector_size(16))) int in
// the device (global) address space; param 2: same pointee in LDS address space.
typedef int v4i_ __attribute__((vector_size(16)));
typedef __attribute__((address_space(1))) v4i_* gv4_t;
typedef __attribute__((address_space(3))) v4i_* lv4_t;
__device__ __forceinline__ void async_cp16(const unsigned short* g, unsigned short* l) {
    __builtin_amdgcn_global_load_async_to_lds_b128((gv4_t)g, (lv4_t)l, 0, 0);
}
__device__ __forceinline__ void wait_async4() { __builtin_amdgcn_s_wait_asynccnt(4); }
__device__ __forceinline__ void wait_async0() { __builtin_amdgcn_s_wait_asynccnt(0); }
#else
__device__ __forceinline__ void wait_async4() {}
__device__ __forceinline__ void wait_async0() {}
#endif

// copy 32 bytes (16 bf16) global -> LDS
__device__ __forceinline__ void stage32B(const unsigned short* g, unsigned short* l) {
#if HAVE_ASYNC_LDS
    async_cp16(g, l);
    async_cp16(g + 8, l + 8);
#else
    ((uint4*)l)[0] = ((const uint4*)g)[0];
    ((uint4*)l)[1] = ((const uint4*)g)[1];
#endif
}

// ---------------------------------------------------------------------------
// Elementwise f32 -> bf16 (x activation), 8 elements/thread.
// ---------------------------------------------------------------------------
__global__ void __launch_bounds__(256)
convert_f32_bf16(const float* __restrict__ src, unsigned short* __restrict__ dst, int n)
{
    int idx = (blockIdx.x * 256 + threadIdx.x) * 8;
    if (idx >= n) return;
    float4 a = ((const float4*)(src + idx))[0];
    float4 b = ((const float4*)(src + idx))[1];
    union { uint4 q; unsigned short s[8]; } pk;
    pk.s[0] = f2bf(a.x); pk.s[1] = f2bf(a.y); pk.s[2] = f2bf(a.z); pk.s[3] = f2bf(a.w);
    pk.s[4] = f2bf(b.x); pk.s[5] = f2bf(b.y); pk.s[6] = f2bf(b.z); pk.s[7] = f2bf(b.w);
    *((uint4*)(dst + idx)) = pk.q;
}

// ---------------------------------------------------------------------------
// Weight transpose+convert: src (K x N) f32 -> dst (N x K) bf16. 32x32 LDS tile.
// ---------------------------------------------------------------------------
__global__ void __launch_bounds__(256)
transpose_convert(const float* __restrict__ src, unsigned short* __restrict__ dst, int K, int N)
{
    __shared__ unsigned short T[32][33];
    int n0 = blockIdx.x * 32, k0 = blockIdx.y * 32;
    int tx = threadIdx.x & 31, ty = threadIdx.x >> 5;  // ty 0..7
#pragma unroll
    for (int r = 0; r < 4; ++r) {
        int k = ty + r * 8;
        T[k][tx] = f2bf(src[(size_t)(k0 + k) * N + n0 + tx]);
    }
    __syncthreads();
#pragma unroll
    for (int r = 0; r < 4; ++r) {
        int n = ty + r * 8;
        dst[(size_t)(n0 + n) * K + k0 + tx] = T[tx][n];
    }
}

// ---------------------------------------------------------------------------
// GEMM: C[M,N] = A[M,K](bf16) * Bt[N,K](bf16, pre-transposed), f32 accum.
// Block tile 128x128x32, 256 threads (8 waves), wave tile 32x64.
// Double-buffered LDS, async global->LDS staging (pure copies).
// mode: 0=f32 out, 1=bf16 out, 2=bf16 + col remap h*64+d -> h*128+d,
//       3=bf16 V^T out: dst[(b*HDIM + col)*LL + l]
// ---------------------------------------------------------------------------
#define BM 128
#define BN 128
#define BK 32
#define LSTR 40   // 32 + 8 pad ushorts; 80B row stride keeps 16B alignment

enum { OUT_F32 = 0, OUT_BF16 = 1, OUT_BF16_REMAP = 2, OUT_BF16_VT = 3 };

__global__ void __launch_bounds__(256)
gemm_bf16_wmma(const unsigned short* __restrict__ A, const unsigned short* __restrict__ Bt,
               void* __restrict__ Cv, int N, int K, int ldo, int mode)
{
    __shared__ unsigned short Alds[2][BM][LSTR];   // [m][k]
    __shared__ unsigned short Blds[2][BN][LSTR];   // [n][k]

    const int tid  = threadIdx.x;
    const int lane = tid & 31, wave = tid >> 5;
    const int hi   = lane >> 4, lm = lane & 15;
    const int wm   = wave >> 1, wn = wave & 1;     // 4x2 wave grid
    const int m0   = blockIdx.y * BM;
    const int n0   = blockIdx.x * BN;

    // staging assignment: 128 rows x 32 elems; each thread 32B
    const int cm = tid >> 1;          // row 0..127
    const int ch = (tid & 1) * 16;    // element offset 0 / 16
    const unsigned short* aSrc = A + (size_t)(m0 + cm) * K + ch;
    const unsigned short* bSrc = Bt + (size_t)(n0 + cm) * K + ch;

    f32x8 acc[2][4];
#pragma unroll
    for (int tm = 0; tm < 2; ++tm)
#pragma unroll
        for (int tn = 0; tn < 4; ++tn) acc[tm][tn] = zero8();

    const int nk = K / BK;
    // prologue: stage tile 0 into buffer 0
    stage32B(aSrc, &Alds[0][cm][ch]);
    stage32B(bSrc, &Blds[0][cm][ch]);

    for (int i = 0; i < nk; ++i) {
        const int cur = i & 1;
        __syncthreads();                                    // prev compute done: safe to overwrite nxt
        if (i + 1 < nk) {
            const int kt = (i + 1) * BK;
            stage32B(aSrc + kt, &Alds[cur ^ 1][cm][ch]);
            stage32B(bSrc + kt, &Blds[cur ^ 1][cm][ch]);
            wait_async4();                                  // cur's 4 copies retired (in-order)
        } else {
            wait_async0();
        }
        __syncthreads();                                    // cur tile visible to all waves

        Frag af[2], bf[4];
#pragma unroll
        for (int tm = 0; tm < 2; ++tm) {
            int m = wm * 32 + tm * 16 + lm;
#pragma unroll
            for (int j = 0; j < 8; ++j)
                af[tm].u32[j] = *(const unsigned int*)&Alds[cur][m][fragA_k(j, hi)];
        }
#pragma unroll
        for (int tn = 0; tn < 4; ++tn) {
            int n = wn * 64 + tn * 16 + lm;
#pragma unroll
            for (int j = 0; j < 8; ++j)
                bf[tn].u32[j] = *(const unsigned int*)&Blds[cur][n][fragB_k(j, hi)];
        }
#pragma unroll
        for (int tm = 0; tm < 2; ++tm)
#pragma unroll
            for (int tn = 0; tn < 4; ++tn)
                acc[tm][tn] = wmma_bf16(af[tm].v, bf[tn].v, acc[tm][tn]);
    }

    // epilogue: D layout lane n=lm, row = r + 8*hi
#pragma unroll
    for (int tm = 0; tm < 2; ++tm)
#pragma unroll
        for (int tn = 0; tn < 4; ++tn) {
            int row0 = m0 + wm * 32 + tm * 16 + hi * 8;
            int col  = n0 + wn * 64 + tn * 16 + lm;
#pragma unroll
            for (int r = 0; r < 8; ++r) {
                int row = row0 + r;
                float v = acc[tm][tn][r];
                if (mode == OUT_F32) {
                    ((float*)Cv)[(size_t)row * ldo + col] = v;
                } else if (mode == OUT_BF16) {
                    ((unsigned short*)Cv)[(size_t)row * ldo + col] = f2bf(v);
                } else if (mode == OUT_BF16_REMAP) {        // k_nope: col = h*64+d -> h*128+d
                    int dc = ((col >> 6) << 7) + (col & 63);
                    ((unsigned short*)Cv)[(size_t)row * ldo + dc] = f2bf(v);
                } else {                                    // V^T: [(b*HDIM + col)][l]
                    size_t dsti = ((size_t)(row >> 11) * HDIM + col) * LL + (row & (LL - 1));
                    ((unsigned short*)Cv)[dsti] = f2bf(v);
                }
            }
        }
}

// ---------------------------------------------------------------------------
// RoPE: pairs (i, i+32) of a 64-wide rope block, angle = pos * 10000^(-i/32).
// dst is always the combined layout: token*2048 + h*128 + 64 + i.
// ---------------------------------------------------------------------------
__global__ void __launch_bounds__(256)
rope_kernel(const unsigned short* __restrict__ src, unsigned short* __restrict__ dst,
            int srcTokStride, int srcHeadStride, int srcOff)
{
    int idx = blockIdx.x * blockDim.x + threadIdx.x;
    if (idx >= NT * NH_ * 32) return;
    int i = idx & 31;
    int h = (idx >> 5) & 15;
    int t = idx >> 9;
    int pos = t & (LL - 1);
    float inv = __expf(-9.210340372f * ((float)i * (1.0f / 32.0f)));  // 10000^(-i/32)
    float ang = (float)pos * inv;
    float c = __cosf(ang), s = __sinf(ang);
    size_t s1 = (size_t)t * srcTokStride + h * srcHeadStride + srcOff + i;
    float x1 = bf2f(src[s1]);
    float x2 = bf2f(src[s1 + 32]);
    size_t d1 = (size_t)t * HDIM + h * HD_ + DNOPE + i;
    dst[d1]      = f2bf(x1 * c - x2 * s);
    dst[d1 + 32] = f2bf(x2 * c + x1 * s);
}

// ---------------------------------------------------------------------------
// Flash attention. Block = 8 waves; each wave owns a 16-query tile (block: 128
// queries of one (b,h)). Per 32-key iteration (double-buffered async staging):
//   S^T (k x q) = K_tile(A) x Q^T(B)    -> softmax per-lane + shfl_xor(16)
//   O^T (d x q) += V^T_tile(A) x P^T(B) -> P^T repacked in-register
// V is pre-transposed in global ((b, h*128+d, l) layout) so staging is a copy.
// ---------------------------------------------------------------------------
__global__ void __launch_bounds__(256)
mla_flash(const unsigned short* __restrict__ Q, const unsigned short* __restrict__ Kp,
          const unsigned short* __restrict__ Vt_g, unsigned short* __restrict__ Ob)
{
    __shared__ unsigned short Klds[2][32][136];  // [k][d], 272B row stride (16B-mult)
    __shared__ unsigned short Vt[2][128][40];    // [d][k], 80B row stride (16B-mult)

    const int tid = threadIdx.x, wave = tid >> 5, lane = tid & 31;
    const int hi = lane >> 4, lm = lane & 15;
    const int qblk = blockIdx.x;           // 0..15
    const int bh   = blockIdx.y;           // 0..31
    const int b = bh >> 4, h = bh & 15;
    const int qr0 = qblk * 128 + wave * 16;
    const size_t base   = (size_t)b * LL * HDIM + (size_t)h * HD_;         // token-major Q/K/O
    const size_t baseVT = ((size_t)b * HDIM + (size_t)h * HD_) * LL;       // V^T

    // staging assignment (per thread, per tile): K 32B + V^T 32B = 4 async ops
    const int kk = tid >> 3, ks = (tid & 7) * 16;   // K: row kk, elems ks..ks+15
    const int dd = tid >> 1, dh = (tid & 1) * 16;   // V^T: row dd, elems dh..dh+15

    // Q^T B-fragments (d x q), 4 chunks of d=32, straight from global (d contiguous)
    Frag qf[4];
    {
        const unsigned short* qrow = Q + base + (size_t)(qr0 + lm) * HDIM;
#pragma unroll
        for (int c = 0; c < 4; ++c)
#pragma unroll
            for (int j = 0; j < 8; ++j)
                qf[c].u32[j] = *(const unsigned int*)(qrow + fragB_k(j, hi) + 32 * c);
    }

    f32x8 o[8];
#pragma unroll
    for (int dt = 0; dt < 8; ++dt) o[dt] = zero8();
    float m_i = NEG_BIG, l_i = 0.f;

    const int nIter = (qblk + 1) * 4;  // keys 0 .. qblk*128+127, 32 per iter

    // prologue: stage keys [0,32) into buffer 0
    stage32B(Kp + base + (size_t)kk * HDIM + ks, &Klds[0][kk][ks]);
    stage32B(Vt_g + baseVT + (size_t)dd * LL + dh, &Vt[0][dd][dh]);

    for (int it = 0; it < nIter; ++it) {
        const int cur = it & 1;
        const int k0 = it * 32;
        __syncthreads();                                    // prev compute done
        if (it + 1 < nIter) {
            const int kn = (it + 1) * 32;
            stage32B(Kp + base + (size_t)(kn + kk) * HDIM + ks, &Klds[cur ^ 1][kk][ks]);
            stage32B(Vt_g + baseVT + (size_t)dd * LL + kn + dh, &Vt[cur ^ 1][dd][dh]);
            wait_async4();
        } else {
            wait_async0();
        }
        __syncthreads();                                    // cur tiles visible

        if (k0 > qr0 + 15) continue;   // wave-uniform; barrier counts still line up

        // S^T: two 16x16 tiles (32 keys), accumulate over d=128
        f32x8 st[2];
#pragma unroll
        for (int t2 = 0; t2 < 2; ++t2) {
            st[t2] = zero8();
            int kr = t2 * 16 + lm;
#pragma unroll
            for (int c = 0; c < 4; ++c) {
                Frag kf;
#pragma unroll
                for (int j = 0; j < 8; ++j)
                    kf.u32[j] = *(const unsigned int*)&Klds[cur][kr][fragA_k(j, hi) + 32 * c];
                st[t2] = wmma_bf16(kf.v, qf[c].v, st[t2]);
            }
        }

        // scale + causal mask + online softmax (lane owns one q column)
        float p[2][8];
        float tmax = NEG_BIG;
        const int qg = qr0 + lm;
#pragma unroll
        for (int t2 = 0; t2 < 2; ++t2)
#pragma unroll
            for (int r = 0; r < 8; ++r) {
                int kg = k0 + t2 * 16 + r + hi * 8;
                float val = st[t2][r] * SCALE;
                if (kg > qg) val = NEG_BIG;
                p[t2][r] = val;
                tmax = fmaxf(tmax, val);
            }
        tmax = fmaxf(tmax, __shfl_xor(tmax, 16, 32));
        float nm = fmaxf(m_i, tmax);
        float alpha = __expf(m_i - nm);
        float rsum = 0.f;
#pragma unroll
        for (int t2 = 0; t2 < 2; ++t2)
#pragma unroll
            for (int r = 0; r < 8; ++r) {
                float e = __expf(p[t2][r] - nm);
                p[t2][r] = e;
                rsum += e;
            }
        rsum += __shfl_xor(rsum, 16, 32);
        l_i = l_i * alpha + rsum;
        m_i = nm;
#pragma unroll
        for (int dt = 0; dt < 8; ++dt) o[dt] = o[dt] * alpha;

        // repack P (D layout of S^T tiles) -> B-operand layout for P^T (32k x 16q):
        // B pair j needs k = 2j + 16*hi; half-swap the 8 k's owned by the other half.
        float x0[8], x1a[8];
#pragma unroll
        for (int r = 0; r < 8; ++r) x0[r] = __shfl_xor(p[0][r], 16, 32);
#pragma unroll
        for (int r = 0; r < 8; ++r) x1a[r] = __shfl_xor(p[1][r], 16, 32);
        Frag pf;
#pragma unroll
        for (int j = 0; j < 4; ++j) {   // k = 2j..2j+1 (+16 if hi)
            float a  = hi ? x1a[2 * j]     : p[0][2 * j];
            float bv = hi ? x1a[2 * j + 1] : p[0][2 * j + 1];
            pf.u32[j] = (unsigned)f2bf(a) | ((unsigned)f2bf(bv) << 16);
        }
#pragma unroll
        for (int j = 0; j < 4; ++j) {   // k = 8+2j (+16 if hi)
            float a  = hi ? p[1][2 * j]     : x0[2 * j];
            float bv = hi ? p[1][2 * j + 1] : x0[2 * j + 1];
            pf.u32[4 + j] = (unsigned)f2bf(a) | ((unsigned)f2bf(bv) << 16);
        }

        // O^T += V^T x P^T over the 8 d-tiles
#pragma unroll
        for (int dt = 0; dt < 8; ++dt) {
            Frag vf;
            int dr = dt * 16 + lm;
#pragma unroll
            for (int j = 0; j < 8; ++j)
                vf.u32[j] = *(const unsigned int*)&Vt[cur][dr][fragA_k(j, hi)];
            o[dt] = wmma_bf16(vf.v, pf.v, o[dt]);
        }
    }

    float inv = 1.0f / l_i;
    unsigned short* orow = Ob + base + (size_t)(qr0 + lm) * HDIM;
#pragma unroll
    for (int dt = 0; dt < 8; ++dt)
#pragma unroll
        for (int r = 0; r < 8; ++r)
            orow[dt * 16 + r + hi * 8] = f2bf(o[dt][r] * inv);
}

// ---------------------------------------------------------------------------
extern "C" void kernel_launch(void* const* d_in, const int* in_sizes, int n_in,
                              void* d_out, int out_size, void* d_ws, size_t ws_size,
                              hipStream_t stream)
{
    (void)in_sizes; (void)n_in; (void)out_size; (void)ws_size;
    const float* x        = (const float*)d_in[0];
    const float* wq       = (const float*)d_in[1];
    const float* wkv_down = (const float*)d_in[2];
    const float* wk_nope  = (const float*)d_in[3];
    const float* wv       = (const float*)d_in[4];
    const float* wk_rope  = (const float*)d_in[5];
    const float* wo       = (const float*)d_in[6];
    float* out = (float*)d_out;

    char* ws = (char*)d_ws;
    size_t off = 0;
    auto take = [&](size_t bytes) -> char* {
        char* p = ws + off;
        off += (bytes + 255) & ~(size_t)255;
        return p;
    };
    unsigned short* xb    = (unsigned short*)take((size_t)NT * HDIM * 2);       // x bf16
    unsigned short* wqT   = (unsigned short*)take((size_t)HDIM * HDIM * 2);
    unsigned short* wkvdT = (unsigned short*)take((size_t)DC * HDIM * 2);
    unsigned short* wknT  = (unsigned short*)take((size_t)1024 * DC * 2);
    unsigned short* wvT   = (unsigned short*)take((size_t)HDIM * DC * 2);
    unsigned short* wkrT  = (unsigned short*)take((size_t)1024 * HDIM * 2);
    unsigned short* woT   = (unsigned short*)take((size_t)HDIM * HDIM * 2);
    unsigned short* qb    = (unsigned short*)take((size_t)NT * HDIM * 2);       // q (rope in place)
    unsigned short* ckv   = (unsigned short*)take((size_t)NT * DC * 2);         // c_kv bf16
    unsigned short* kb    = (unsigned short*)take((size_t)NT * HDIM * 2);       // combined K [nope|rope]
    unsigned short* kr    = (unsigned short*)take((size_t)NT * NH_ * DROPE * 2);// raw k_rope
    unsigned short* vt    = (unsigned short*)take((size_t)NT * HDIM * 2);       // V^T (b, h*128+d, l)
    unsigned short* ob    = (unsigned short*)take((size_t)NT * HDIM * 2);       // attention out bf16

    dim3 blk(256);
    // one-time precision/layout prep (bandwidth-trivial vs. the matmul work)
    convert_f32_bf16<<<(NT * HDIM) / (256 * 8), blk, 0, stream>>>(x, xb, NT * HDIM);
    transpose_convert<<<dim3(HDIM / 32, HDIM / 32), blk, 0, stream>>>(wq, wqT, HDIM, HDIM);
    transpose_convert<<<dim3(DC / 32, HDIM / 32), blk, 0, stream>>>(wkv_down, wkvdT, HDIM, DC);
    transpose_convert<<<dim3(1024 / 32, DC / 32), blk, 0, stream>>>(wk_nope, wknT, DC, 1024);
    transpose_convert<<<dim3(HDIM / 32, DC / 32), blk, 0, stream>>>(wv, wvT, DC, HDIM);
    transpose_convert<<<dim3(1024 / 32, HDIM / 32), blk, 0, stream>>>(wk_rope, wkrT, HDIM, 1024);
    transpose_convert<<<dim3(HDIM / 32, HDIM / 32), blk, 0, stream>>>(wo, woT, HDIM, HDIM);

    // q = x @ wq                  (4096 x 2048 x 2048) -> bf16
    gemm_bf16_wmma<<<dim3(HDIM / BN, NT / BM), blk, 0, stream>>>(xb, wqT, qb, HDIM, HDIM, HDIM, OUT_BF16);
    // c_kv = x @ wkv_down         (4096 x 512 x 2048) -> bf16
    gemm_bf16_wmma<<<dim3(DC / BN, NT / BM), blk, 0, stream>>>(xb, wkvdT, ckv, DC, HDIM, DC, OUT_BF16);
    // k_nope = c_kv @ wk_nope     (4096 x 1024 x 512) -> kb (col remap h*64+d -> h*128+d)
    gemm_bf16_wmma<<<dim3(1024 / BN, NT / BM), blk, 0, stream>>>(ckv, wknT, kb, 1024, DC, HDIM, OUT_BF16_REMAP);
    // k_rope = x @ wk_rope        (4096 x 1024 x 2048) -> kr
    gemm_bf16_wmma<<<dim3(1024 / BN, NT / BM), blk, 0, stream>>>(xb, wkrT, kr, 1024, HDIM, 1024, OUT_BF16);
    // v = c_kv @ wv               (4096 x 2048 x 512) -> V^T layout
    gemm_bf16_wmma<<<dim3(HDIM / BN, NT / BM), blk, 0, stream>>>(ckv, wvT, vt, HDIM, DC, 0, OUT_BF16_VT);

    // RoPE: q in place; kr -> rope half of kb
    int rthreads = NT * NH_ * 32;
    rope_kernel<<<(rthreads + 255) / 256, blk, 0, stream>>>(qb, qb, HDIM, HD_, DNOPE);
    rope_kernel<<<(rthreads + 255) / 256, blk, 0, stream>>>(kr, kb, NH_ * DROPE, DROPE, 0);

    // causal flash attention -> ob (bf16)
    mla_flash<<<dim3(LL / 128, BB * NH_), blk, 0, stream>>>(qb, kb, vt, ob);

    // out = ob @ wo               (4096 x 2048 x 2048) -> f32
    gemm_bf16_wmma<<<dim3(HDIM / BN, NT / BM), blk, 0, stream>>>(ob, woT, out, HDIM, HDIM, HDIM, OUT_F32);
}
